// CrossAttention_40295383171353
// MI455X (gfx1250) — matrix operand order repaired and checked by reading, not compiled, not run on hardware
//
#include <hip/hip_runtime.h>
#include <hip/hip_bf16.h>

typedef __bf16 bf16;
typedef __attribute__((ext_vector_type(16))) __bf16 v16bf;
typedef __attribute__((ext_vector_type(8)))  __bf16 v8bf;
typedef __attribute__((ext_vector_type(8)))  float  v8f;
typedef __attribute__((ext_vector_type(4)))  float  v4f;

typedef unsigned int u32;
typedef __attribute__((ext_vector_type(4))) unsigned int u32x4;
typedef __attribute__((ext_vector_type(8))) int i32x8;
typedef __attribute__((ext_vector_type(4))) int i32x4;

#define WMMA_BF16(A,B,C) __builtin_amdgcn_wmma_f32_16x16x32_bf16( \
    false, (A), false, (B), (short)0, (C), false, false)

static __device__ __forceinline__ v16bf join16(v8bf lo, v8bf hi) {
  v16bf r;
#pragma unroll
  for (int e = 0; e < 8; ++e) { r[e] = lo[e]; r[e + 8] = hi[e]; }
  return r;
}

// ---------------------------------------------------------------------------
// TDM: DMA one 2-D bf16 tile (tile_rows x tile_cols, row stride in elements)
// from global memory into LDS (packed row-major at lds_off).
// D# packing per CDNA5 ISA:
//   group0: [1:0]=count=1, [63:32]=lds_addr, [120:64]=global_addr, [127:126]=2
//   group1: [17:16]=data_size(1->2B), [79:48]=tensor_dim0, [111:80]=tensor_dim1,
//           [127:112]=tile_dim0, [143:128]=tile_dim1, [207:160]=dim0_stride
// Tensor dims sized so the in-bounds tile never trips OOB handling.
// This toolchain exposes the 6-arg builtin: (g0, g1, g2, g3, g4, cpol).
// ---------------------------------------------------------------------------
static __device__ __forceinline__ void tdm_load_tile_2d(
    unsigned lds_off, const bf16* gsrc, u32 tile_cols, u32 tile_rows,
    u32 row_stride_elems) {
  unsigned long long ga = (unsigned long long)(const void*)gsrc;
  u32x4 g0;
  g0[0] = 1u;                                   // count=1, user descriptor
  g0[1] = lds_off;                              // lds_addr (bytes)
  g0[2] = (u32)ga;                              // global_addr[31:0]
  g0[3] = ((u32)(ga >> 32) & 0x01FFFFFFu) | (2u << 30);  // addr[56:32], type=2
  const u32 td0 = row_stride_elems;             // tensor_dim0 >= tile_cols
  const u32 td1 = 0x40000000u;                  // effectively unbounded rows
  i32x8 g1;
  g1[0] = (int)(1u << 16);                      // wg_mask=0, data_size=1 (2B)
  g1[1] = (int)((td0 & 0xFFFFu) << 16);         // tensor_dim0[15:0]
  g1[2] = (int)((td0 >> 16) | ((td1 & 0xFFFFu) << 16));
  g1[3] = (int)((td1 >> 16) | (tile_cols << 16));       // tile_dim0
  g1[4] = (int)tile_rows;                       // tile_dim1; tile_dim2=0
  g1[5] = (int)row_stride_elems;                // tensor_dim0_stride[31:0]
  g1[6] = 0;
  g1[7] = 0;
  i32x4 z4 = {};
  i32x8 z8 = {};
  __builtin_amdgcn_tensor_load_to_lds(g0, g1, z4, z4, z8, 0);
}

// ---------------------------------------------------------------------------
// Kernel 1: QKV projection. C = X(f32)[4096,1024] @ W(f32)[1024,1024] -> bf16
// blockIdx.z in {0,1,2} selects (Wq->Q, Wk->K, Wv->V).
// Block tile 128x128, BK=64; 8 waves, each 32x64 (2x4 WMMA tiles).
// ---------------------------------------------------------------------------
__global__ __launch_bounds__(256) void qkv_gemm(
    const float* __restrict__ X,
    const float* __restrict__ Wq, const float* __restrict__ Wk,
    const float* __restrict__ Wv,
    bf16* __restrict__ Q, bf16* __restrict__ Kd, bf16* __restrict__ V) {
  constexpr int KD = 1024, ND = 1024;
  const float* W = (blockIdx.z == 0) ? Wq : (blockIdx.z == 1) ? Wk : Wv;
  bf16* C = (blockIdx.z == 0) ? Q : (blockIdx.z == 1) ? Kd : V;

  __shared__ __align__(32) bf16 As[128 * 64];   // [m][k]
  __shared__ __align__(32) bf16 Bs[64 * 128];   // [k][n]

  const int tid = threadIdx.x, lane = tid & 31, wid = tid >> 5;
  const int half = lane >> 4, l15 = lane & 15;
  const int wm = wid >> 1, wn = wid & 1;            // 4x2 wave grid
  const int m0 = blockIdx.y * 128, n0 = blockIdx.x * 128;

  v8f acc[2][4] = {};

  for (int kt = 0; kt < KD; kt += 64) {
    __syncthreads();
    // stage A tile (f32 -> bf16): 128x64 = 2048 float4 chunks
#pragma unroll
    for (int i = 0; i < 8; ++i) {
      int c = tid + i * 256;
      int r = c >> 4, cc = (c & 15) << 2;
      v4f a = *(const v4f*)(X + (size_t)(m0 + r) * KD + kt + cc);
      bf16* d = &As[r * 64 + cc];
      d[0] = (bf16)a[0]; d[1] = (bf16)a[1]; d[2] = (bf16)a[2]; d[3] = (bf16)a[3];
    }
    // stage B tile (f32 -> bf16): 64x128
#pragma unroll
    for (int i = 0; i < 8; ++i) {
      int c = tid + i * 256;
      int r = c >> 5, cc = (c & 31) << 2;
      v4f b = *(const v4f*)(W + (size_t)(kt + r) * ND + n0 + cc);
      bf16* d = &Bs[r * 128 + cc];
      d[0] = (bf16)b[0]; d[1] = (bf16)b[1]; d[2] = (bf16)b[2]; d[3] = (bf16)b[3];
    }
    __syncthreads();

#pragma unroll
    for (int kk = 0; kk < 2; ++kk) {
      v16bf af[2];
#pragma unroll
      for (int mt = 0; mt < 2; ++mt) {
        const bf16* p = &As[(wm * 32 + mt * 16 + l15) * 64 + kk * 32 + half * 8];
        af[mt] = join16(*(const v8bf*)p, *(const v8bf*)(p + 16));
      }
#pragma unroll
      for (int nt = 0; nt < 4; ++nt) {
        const bf16* p = &Bs[(kk * 32 + half * 8) * 128 + wn * 64 + nt * 16 + l15];
        v16bf bfr;
#pragma unroll
        for (int e = 0; e < 16; ++e) bfr[e] = p[((e & 7) + (e >> 3) * 16) * 128];
#pragma unroll
        for (int mt = 0; mt < 2; ++mt)
          acc[mt][nt] = WMMA_BF16(af[mt], bfr, acc[mt][nt]);
      }
    }
  }

  // epilogue: C/D layout: VGPR j -> row j + half*8, lane&15 -> col
#pragma unroll
  for (int mt = 0; mt < 2; ++mt)
#pragma unroll
    for (int nt = 0; nt < 4; ++nt)
#pragma unroll
      for (int j = 0; j < 8; ++j) {
        int row = m0 + wm * 32 + mt * 16 + j + half * 8;
        int col = n0 + wn * 64 + nt * 16 + l15;
        C[(size_t)row * ND + col] = (bf16)acc[mt][nt][j];
      }
}

// ---------------------------------------------------------------------------
// Kernel 2: flash attention per (b, h, 128 query rows). Online softmax.
// 8 waves x 16 q-rows. KV tile 128. K staged transposed in LDS by the VALU;
// V staged by the Tensor Data Mover (wave 0 issues, TENSORcnt drained before
// the barrier). The P matrix reuses the K^T buffer (barrier-protected).
// ---------------------------------------------------------------------------
__global__ __launch_bounds__(256) void attn_kernel(
    const bf16* __restrict__ Q, const bf16* __restrict__ Kg,
    const bf16* __restrict__ Vg, bf16* __restrict__ O) {
  constexpr int S = 2048, HD = 64, ND = 1024;
  const float SCALE = 0.125f;  // 64^-0.5

  __shared__ __align__(32) bf16 KtPs[128 * 128];  // K^T [64][128] / P [128][128]
  __shared__ __align__(32) bf16 Vs[128 * 64];     // V   [kv][d]

  const int tid = threadIdx.x, lane = tid & 31, wid = tid >> 5;
  const int half = lane >> 4, l15 = lane & 15;
  const int h = blockIdx.y, b = blockIdx.z;
  const int qbase = b * S + blockIdx.x * 128;

  // Q fragments (wave's 16 rows x d=64) stay in registers for the whole loop
  const int qrow = qbase + wid * 16 + l15;
  v16bf qfrag[2];
#pragma unroll
  for (int kk = 0; kk < 2; ++kk) {
    const bf16* p = Q + (size_t)qrow * ND + h * HD + kk * 32 + half * 8;
    qfrag[kk] = join16(*(const v8bf*)p, *(const v8bf*)(p + 16));
  }

  v8f accO[4] = {};
  float m[8], l[8];
#pragma unroll
  for (int j = 0; j < 8; ++j) { m[j] = -3.0e38f; l[j] = 0.f; }

  const unsigned vs_off = (unsigned)(unsigned long long)(void*)Vs;

  for (int it = 0; it < S / 128; ++it) {
    __syncthreads();  // previous iteration done reading P/V buffers
    const int kbase = b * S + it * 128;

    // V tile via Tensor Data Mover: 128 rows x 64 bf16, row stride 1024
    if (wid == 0)
      tdm_load_tile_2d(vs_off, Vg + (size_t)kbase * ND + h * HD, 64, 128, ND);

    // stage K^T with the VALU: 128 rows x 64 d, 8-elem (16B) chunks
#pragma unroll
    for (int i = 0; i < 4; ++i) {
      int c = tid + i * 256;
      int r = c >> 3, dc = (c & 7) << 3;
      v8bf kv = *(const v8bf*)(Kg + (size_t)(kbase + r) * ND + h * HD + dc);
#pragma unroll
      for (int e = 0; e < 8; ++e) KtPs[(dc + e) * 128 + r] = kv[e];
    }
    if (wid == 0) __builtin_amdgcn_s_wait_tensorcnt(0);  // V DMA done
    __syncthreads();

    // scores: wave's 16 q-rows x 128 kv-cols
    v8f accS[8] = {};
#pragma unroll
    for (int kk = 0; kk < 2; ++kk) {
#pragma unroll
      for (int n = 0; n < 8; ++n) {
        const bf16* p = &KtPs[(kk * 32 + half * 8) * 128 + n * 16 + l15];
        v16bf bfr;
#pragma unroll
        for (int e = 0; e < 16; ++e) bfr[e] = p[((e & 7) + (e >> 3) * 16) * 128];
        accS[n] = WMMA_BF16(qfrag[kk], bfr, accS[n]);
      }
    }
    __syncthreads();  // all waves done reading K^T; P may overwrite it

    // online softmax; write P (bf16) into the shared buffer in A-layout rows
#pragma unroll
    for (int j = 0; j < 8; ++j) {
      float rm = -3.0e38f;
#pragma unroll
      for (int n = 0; n < 8; ++n) rm = fmaxf(rm, accS[n][j] * SCALE);
#pragma unroll
      for (int mk = 1; mk < 16; mk <<= 1) rm = fmaxf(rm, __shfl_xor(rm, mk, 32));
      float mn = fmaxf(m[j], rm);
      float alpha = __expf(m[j] - mn);
      float ps = 0.f;
#pragma unroll
      for (int n = 0; n < 8; ++n) {
        float p = __expf(accS[n][j] * SCALE - mn);
        ps += p;
        KtPs[(wid * 16 + j + half * 8) * 128 + n * 16 + l15] = (bf16)p;
      }
#pragma unroll
      for (int mk = 1; mk < 16; mk <<= 1) ps += __shfl_xor(ps, mk, 32);
      l[j] = l[j] * alpha + ps;
      m[j] = mn;
#pragma unroll
      for (int o = 0; o < 4; ++o) accO[o][j] *= alpha;
    }
    __syncthreads();  // P visible

    // O += P @ V : k over 128 kv in 4 steps of 32
#pragma unroll
    for (int kk = 0; kk < 4; ++kk) {
      const bf16* pa = &KtPs[(wid * 16 + l15) * 128 + kk * 32 + half * 8];
      v16bf af = join16(*(const v8bf*)pa, *(const v8bf*)(pa + 16));
#pragma unroll
      for (int o = 0; o < 4; ++o) {
        const bf16* pb = &Vs[(kk * 32 + half * 8) * 64 + o * 16 + l15];
        v16bf bfr;
#pragma unroll
        for (int e = 0; e < 16; ++e) bfr[e] = pb[((e & 7) + (e >> 3) * 16) * 64];
        accO[o] = WMMA_BF16(af, bfr, accO[o]);
      }
    }
  }

  // normalize and store attn output bf16 at [row][h*64+d]
#pragma unroll
  for (int o = 0; o < 4; ++o)
#pragma unroll
    for (int j = 0; j < 8; ++j) {
      int row = qbase + wid * 16 + j + half * 8;
      int col = h * HD + o * 16 + l15;
      O[(size_t)row * ND + col] = (bf16)(accO[o][j] / l[j]);
    }
}

// ---------------------------------------------------------------------------
// Kernel 3: output projection. C = attn(bf16)[4096,1024] @ Wo(f32) + b_o -> f32
// A tile is a verbatim bf16 copy -> staged by the Tensor Data Mover.
// ---------------------------------------------------------------------------
__global__ __launch_bounds__(256) void out_gemm(
    const bf16* __restrict__ A, const float* __restrict__ W,
    const float* __restrict__ bias, float* __restrict__ C) {
  constexpr int KD = 1024, ND = 1024;
  __shared__ __align__(32) bf16 As[128 * 64];
  __shared__ __align__(32) bf16 Bs[64 * 128];

  const int tid = threadIdx.x, lane = tid & 31, wid = tid >> 5;
  const int half = lane >> 4, l15 = lane & 15;
  const int wm = wid >> 1, wn = wid & 1;
  const int m0 = blockIdx.y * 128, n0 = blockIdx.x * 128;

  const unsigned as_off = (unsigned)(unsigned long long)(void*)As;

  v8f acc[2][4] = {};

  for (int kt = 0; kt < KD; kt += 64) {
    __syncthreads();
    // A tile (bf16, 128x64, row stride 1024) via TDM
    if (wid == 0)
      tdm_load_tile_2d(as_off, A + (size_t)m0 * KD + kt, 64, 128, KD);
    // W f32 -> bf16 staged by the VALU
#pragma unroll
    for (int i = 0; i < 8; ++i) {
      int c = tid + i * 256;
      int r = c >> 5, cc = (c & 31) << 2;
      v4f bv = *(const v4f*)(W + (size_t)(kt + r) * ND + n0 + cc);
      bf16* d = &Bs[r * 128 + cc];
      d[0] = (bf16)bv[0]; d[1] = (bf16)bv[1]; d[2] = (bf16)bv[2]; d[3] = (bf16)bv[3];
    }
    if (wid == 0) __builtin_amdgcn_s_wait_tensorcnt(0);
    __syncthreads();

#pragma unroll
    for (int kk = 0; kk < 2; ++kk) {
      v16bf af[2];
#pragma unroll
      for (int mt = 0; mt < 2; ++mt) {
        const bf16* p = &As[(wm * 32 + mt * 16 + l15) * 64 + kk * 32 + half * 8];
        af[mt] = join16(*(const v8bf*)p, *(const v8bf*)(p + 16));
      }
#pragma unroll
      for (int nt = 0; nt < 4; ++nt) {
        const bf16* p = &Bs[(kk * 32 + half * 8) * 128 + wn * 64 + nt * 16 + l15];
        v16bf bfr;
#pragma unroll
        for (int e = 0; e < 16; ++e) bfr[e] = p[((e & 7) + (e >> 3) * 16) * 128];
#pragma unroll
        for (int mt = 0; mt < 2; ++mt)
          acc[mt][nt] = WMMA_BF16(af[mt], bfr, acc[mt][nt]);
      }
    }
  }

#pragma unroll
  for (int mt = 0; mt < 2; ++mt)
#pragma unroll
    for (int nt = 0; nt < 4; ++nt)
#pragma unroll
      for (int j = 0; j < 8; ++j) {
        int row = m0 + wm * 32 + mt * 16 + j + half * 8;
        int col = n0 + wn * 64 + nt * 16 + l15;
        C[(size_t)row * ND + col] = acc[mt][nt][j] + bias[col];
      }
}

// ---------------------------------------------------------------------------
extern "C" void kernel_launch(void* const* d_in, const int* in_sizes, int n_in,
                              void* d_out, int out_size, void* d_ws,
                              size_t ws_size, hipStream_t stream) {
  const float* x  = (const float*)d_in[0];
  const float* wq = (const float*)d_in[1];
  const float* wk = (const float*)d_in[2];
  const float* wv = (const float*)d_in[3];
  const float* wo = (const float*)d_in[4];
  const float* bo = (const float*)d_in[5];

  const size_t MAT = (size_t)4096 * 1024;  // B*S x inner_dim
  bf16* q = (bf16*)d_ws;
  bf16* k = q + MAT;
  bf16* v = k + MAT;
  bf16* a = v + MAT;  // total 32 MB bf16 workspace

  qkv_gemm<<<dim3(8, 32, 3), 256, 0, stream>>>(x, wq, wk, wv, q, k, v);
  attn_kernel<<<dim3(16, 16, 2), 256, 0, stream>>>(q, k, v, a);
  out_gemm<<<dim3(8, 32, 1), 256, 0, stream>>>(a, wo, bo, (float*)d_out);
}
